// MapEncoder_76441827934897
// MI455X (gfx1250) — matrix operand order repaired and checked by reading, not compiled, not run on hardware
//
#include <hip/hip_runtime.h>
#include <hip/hip_bf16.h>

#define N_NODES 100000
#define N_EDGES 100000
#define EPS 1e-5f

typedef __bf16 bf16;
typedef __attribute__((ext_vector_type(16))) __bf16 v16bf;
typedef __attribute__((ext_vector_type(8)))  float  v8f;
typedef __attribute__((ext_vector_type(4)))  unsigned int u32x4;

union Frag16 { v16bf v; u32x4 q[2]; };

// ---------------------------------------------------------------------------
// Wave-level 16xKa (A, LDS) x Kax32 (two 16-wide B tiles of W^T) -> two 16x16
// f32 accumulators. A fragment is loaded ONCE per K-chunk and reused by both
// WMMAs (halves ds_load_b128 traffic vs. per-tile reload).
// Fragment layouts per CDNA5 ISA 7.12.2 (wave32):
//   A 16x32 bf16: lane L: M=L&15, half=L>>4; elems j<8 -> K=32kc+8*half+j,
//                 j>=8 -> K=32kc+16+8*half+(j-8)   (two 16B LDS loads)
//   B 32x16 bf16: lane L: N=L&15, half=L>>4; same K mapping as A
// ---------------------------------------------------------------------------
__device__ __forceinline__ void wave_gemm_bf16_x2(
    const bf16* __restrict__ sA, int Ka,
    const bf16* __restrict__ W, int lane, int ot0, int ot1,
    v8f& acc0, v8f& acc1)
{
    const int m    = lane & 15;
    const int half = lane >> 4;
    const int chunks = Ka >> 5;
    for (int kc = 0; kc < chunks; ++kc) {
        Frag16 a, b0, b1;
        const bf16* rA = sA + m * Ka + kc * 32;
        a.q[0] = *(const u32x4*)(rA + half * 8);
        a.q[1] = *(const u32x4*)(rA + 16 + half * 8);
        const bf16* rB0 = W + (size_t)(ot0 * 16 + m) * Ka + kc * 32;
        const bf16* rB1 = W + (size_t)(ot1 * 16 + m) * Ka + kc * 32;
        b0.q[0] = *(const u32x4*)(rB0 + half * 8);
        b0.q[1] = *(const u32x4*)(rB0 + 16 + half * 8);
        b1.q[0] = *(const u32x4*)(rB1 + half * 8);
        b1.q[1] = *(const u32x4*)(rB1 + 16 + half * 8);
        acc0 = __builtin_amdgcn_wmma_f32_16x16x32_bf16(false, a.v, false, b0.v,
                                                       (short)0, acc0, false, false);
        acc1 = __builtin_amdgcn_wmma_f32_16x16x32_bf16(false, a.v, false, b1.v,
                                                       (short)0, acc1, false, false);
    }
}

// ---------------------------------------------------------------------------
// Fused GEMM (+ GroupNorm(1) + post-ops). Block = 128 thr = 4 waves, 16 rows.
// mode 0: outF = A@W^T                         (ctr pre-accumulator)
// mode 1: outF = gn(A@W^T)                     (input branch)
// mode 2: outB = bf16(relu(gn(A@W^T)+addsrc))  (seg branch -> meta input)
// mode 3: v=relu(gn(..)); outF,outB,resOut=v   (meta)
// mode 4: v=relu(gn(..)+addsrc); outF,outB,resOut=v   (ctr2 + residual)
// ---------------------------------------------------------------------------
__global__ __launch_bounds__(128) void k_gemm16(
    const bf16* __restrict__ A, int Ka,
    const bf16* __restrict__ W,
    const float* __restrict__ g, const float* __restrict__ bg,
    const float* __restrict__ addsrc,
    float* __restrict__ outF,
    bf16* __restrict__ outB, int outBstride,
    float* __restrict__ resOut,
    int mode)
{
    __shared__ __align__(16) bf16  sA[16 * 160];
    __shared__ __align__(16) float sOut[16 * 132];   // stride 132: no bank conflicts
    __shared__ float sMean[16], sRstd[16];

    const int tid  = threadIdx.x;
    const size_t row0 = (size_t)blockIdx.x * 16;

    { // A tile is 16 consecutive rows -> contiguous 16*Ka bf16 in global
        const u32x4* src = (const u32x4*)(A + row0 * Ka);
        u32x4* dst = (u32x4*)sA;
        const int nvec = (16 * Ka) >> 3;
        for (int i = tid; i < nvec; i += 128) dst[i] = src[i];
    }
    __syncthreads();

    const int lane = tid & 31, wave = tid >> 5;
    const int m = lane & 15, half = lane >> 4;
    {
        v8f acc0 = {}, acc1 = {};
        wave_gemm_bf16_x2(sA, Ka, W, lane, wave, wave + 4, acc0, acc1);
#pragma unroll
        for (int r = 0; r < 8; ++r) {
            sOut[(r + 8 * half) * 132 + wave * 16 + m]       = acc0[r];
            sOut[(r + 8 * half) * 132 + (wave + 4) * 16 + m] = acc1[r];
        }
    }
    __syncthreads();

    if (tid < 16 && mode != 0) {
        float s = 0.f, s2 = 0.f;
        for (int c = 0; c < 128; ++c) { float v = sOut[tid * 132 + c]; s += v; s2 += v * v; }
        float mean = s * (1.f / 128.f);
        float var  = s2 * (1.f / 128.f) - mean * mean;
        sMean[tid] = mean;
        sRstd[tid] = rsqrtf(var + EPS);
    }
    __syncthreads();

    const int rr = tid >> 3, c0 = (tid & 7) * 16;
    const size_t rowg = row0 + rr;
    float mean = 0.f, rs = 1.f;
    if (mode != 0) { mean = sMean[rr]; rs = sRstd[rr]; }
    for (int j = 0; j < 16; ++j) {
        const int c = c0 + j;
        float v = sOut[rr * 132 + c];
        if (mode != 0)               v = (v - mean) * rs * g[c] + bg[c];
        if (mode == 2 || mode == 4)  v += addsrc[rowg * 128 + c];
        if (mode >= 2)               v = fmaxf(v, 0.f);
        if (mode != 2 && outF)       outF[rowg * 128 + c] = v;
        if (mode >= 2 && outB)       outB[rowg * (size_t)outBstride + c] = (bf16)v;
        if ((mode == 3 || mode == 4) && resOut) resOut[rowg * 128 + c] = v;
    }
}

// ---------------------------------------------------------------------------
// Edge message passing: block = (16 edges) x (edge type t).
// Gather feat_bf16[src] rows -> LDS, WMMA against edge_W[i,t], atomic-scatter
// f32 into temp[dst] (agent-scope relaxed -> global_atomic_add_f32).
// ---------------------------------------------------------------------------
__global__ __launch_bounds__(128) void k_edges(
    const bf16* __restrict__ FB, const bf16* __restrict__ EW,
    const int* __restrict__ indexes, const int* __restrict__ mask,
    float* __restrict__ T)
{
    __shared__ __align__(16) bf16 sA[16 * 128];
    __shared__ int sSrc[16], sDst[16], sVal[16];

    const int t  = blockIdx.y;
    const int e0 = blockIdx.x * 16;
    const int mt = mask[t];
    if (e0 >= mt) return;                       // block-uniform

    const int tid = threadIdx.x;
    if (tid < 16) {
        const int e = e0 + tid;
        const int v = (e < mt) ? 1 : 0;
        sVal[tid] = v;
        sSrc[tid] = v ? indexes[(size_t)e * 28 + 2 * t + 1] : 0;
        sDst[tid] = v ? indexes[(size_t)e * 28 + 2 * t]     : 0;
    }
    __syncthreads();

    for (int i = tid; i < 256; i += 128) {      // 16 rows x 16 u32x4
        const int r = i >> 4, p = i & 15;
        u32x4 val = {0u, 0u, 0u, 0u};
        if (sVal[r]) val = ((const u32x4*)(FB + (size_t)sSrc[r] * 128))[p];
        ((u32x4*)sA)[i] = val;
    }
    __syncthreads();

    const int lane = tid & 31, wave = tid >> 5;
    const int m = lane & 15, half = lane >> 4;
    const bf16* W = EW + (size_t)t * (128 * 128);
    v8f acc0 = {}, acc1 = {};
    wave_gemm_bf16_x2(sA, 128, W, lane, wave, wave + 4, acc0, acc1);
#pragma unroll
    for (int r = 0; r < 8; ++r) {
        const int row = r + 8 * half;
        if (sVal[row]) {
            float* p0 = T + (size_t)sDst[row] * 128 + wave * 16 + m;
            __hip_atomic_fetch_add(p0, acc0[r], __ATOMIC_RELAXED,
                                   __HIP_MEMORY_SCOPE_AGENT);
            float* p1 = T + (size_t)sDst[row] * 128 + (wave + 4) * 16 + m;
            __hip_atomic_fetch_add(p1, acc1[r], __ATOMIC_RELAXED,
                                   __HIP_MEMORY_SCOPE_AGENT);
        }
    }
}

// ---------------------------------------------------------------------------
// Row GroupNorm(1) + ReLU on temp, emit bf16 (input of ctr2 GEMM).
// ---------------------------------------------------------------------------
__global__ __launch_bounds__(128) void k_rownorm_relu(
    const float* __restrict__ T, const float* __restrict__ g,
    const float* __restrict__ bg, bf16* __restrict__ out)
{
    __shared__ __align__(16) float sT[16 * 132];
    __shared__ float sMean[16], sRstd[16];
    const int tid = threadIdx.x;
    const size_t row0 = (size_t)blockIdx.x * 16;
    for (int i = tid; i < 16 * 32; i += 128) {
        const int r = i >> 5, p = i & 31;
        ((u32x4*)(sT + r * 132))[p] = ((const u32x4*)(T + (row0 + r) * 128))[p];
    }
    __syncthreads();
    if (tid < 16) {
        float s = 0.f, s2 = 0.f;
        for (int c = 0; c < 128; ++c) { float v = sT[tid * 132 + c]; s += v; s2 += v * v; }
        float mean = s * (1.f / 128.f);
        float var  = s2 * (1.f / 128.f) - mean * mean;
        sMean[tid] = mean; sRstd[tid] = rsqrtf(var + EPS);
    }
    __syncthreads();
    const int r = tid >> 3, c0 = (tid & 7) * 16;
    const float mean = sMean[r], rs = sRstd[r];
    for (int j = 0; j < 16; ++j) {
        const int c = c0 + j;
        const float v = (sT[r * 132 + c] - mean) * rs * g[c] + bg[c];
        out[(row0 + r) * 128 + c] = (bf16)fmaxf(v, 0.f);
    }
}

// --------------------- small elementwise helper kernels --------------------
__global__ void k_lin2relu(const float* __restrict__ nodes,
                           const float* __restrict__ W1, const float* __restrict__ b1,
                           bf16* __restrict__ out, int col0)
{
    const int idx = blockIdx.x * 256 + threadIdx.x;
    if (idx >= N_NODES * 128) return;
    const int n = idx >> 7, c = idx & 127;
    const float x0 = nodes[(size_t)n * 8 + col0];
    const float x1 = nodes[(size_t)n * 8 + col0 + 1];
    const float v = x0 * W1[c * 2] + x1 * W1[c * 2 + 1] + b1[c];
    out[idx] = (bf16)fmaxf(v, 0.f);
}

__global__ void k_meta_extra(const float* __restrict__ nodes, bf16* __restrict__ MB)
{
    const int idx = blockIdx.x * 256 + threadIdx.x;
    if (idx >= N_NODES * 32) return;
    const int n = idx >> 5, c = 128 + (idx & 31);
    float v = 0.f;
    if (c < 132) v = nodes[(size_t)n * 8 + 4 + (c - 128)];
    MB[(size_t)n * 160 + c] = (bf16)v;
}

__global__ void k_cvt(const float* __restrict__ src, bf16* __restrict__ dst, int n)
{
    const int idx = blockIdx.x * 256 + threadIdx.x;
    if (idx < n) dst[idx] = (bf16)src[idx];
}

__global__ void k_cvt_meta(const float* __restrict__ src, bf16* __restrict__ dst)
{
    const int idx = blockIdx.x * 256 + threadIdx.x;   // 128 x 160 padded
    if (idx >= 128 * 160) return;
    const int o = idx / 160, c = idx % 160;
    dst[idx] = (c < 132) ? (bf16)src[o * 132 + c] : (bf16)0.f;
}

__global__ void k_copy_xy(const float* __restrict__ nodes, float* __restrict__ out)
{
    const int idx = blockIdx.x * 256 + threadIdx.x;
    if (idx >= N_NODES * 2) return;
    const int n = idx >> 1, j = idx & 1;
    out[idx] = nodes[(size_t)n * 8 + j];
}

// ---------------------------------------------------------------------------
extern "C" void kernel_launch(void* const* d_in, const int* in_sizes, int n_in,
                              void* d_out, int out_size, void* d_ws, size_t ws_size,
                              hipStream_t stream)
{
    (void)in_sizes; (void)n_in; (void)out_size;
    const float* nodes   = (const float*)d_in[0];
    const int*   indexes = (const int*)  d_in[1];
    const int*   mask    = (const int*)  d_in[2];
    const float* in1_W   = (const float*)d_in[3];
    const float* in1_b   = (const float*)d_in[4];
    const float* in2_W   = (const float*)d_in[5];
    const float* in_g    = (const float*)d_in[6];
    const float* in_bg   = (const float*)d_in[7];
    const float* seg1_W  = (const float*)d_in[8];
    const float* seg1_b  = (const float*)d_in[9];
    const float* seg2_W  = (const float*)d_in[10];
    const float* seg_g   = (const float*)d_in[11];
    const float* seg_bg  = (const float*)d_in[12];
    const float* meta_W  = (const float*)d_in[13];
    const float* meta_g  = (const float*)d_in[14];
    const float* meta_bg = (const float*)d_in[15];
    const float* ctr_W   = (const float*)d_in[16];
    const float* edge_W  = (const float*)d_in[17];
    const float* norm_g  = (const float*)d_in[18];
    const float* norm_bg = (const float*)d_in[19];
    const float* ctr2_W  = (const float*)d_in[20];
    const float* ctr2_g  = (const float*)d_in[21];
    const float* ctr2_bg = (const float*)d_in[22];
    float* out = (float*)d_out;

    // ---- workspace carve-out (~240 MB) ----
    char* ws = (char*)d_ws;
    size_t off = 0;
    auto carve = [&](size_t bytes) -> char* {
        char* p = ws + off; off += (bytes + 255) & ~(size_t)255; return p;
    };
    float* F   = (float*)carve((size_t)N_NODES * 128 * 4);   // feat f32
    float* R   = (float*)carve((size_t)N_NODES * 128 * 4);   // residual f32
    float* T   = (float*)carve((size_t)N_NODES * 128 * 4);   // temp accumulator
    bf16*  FB  = (bf16*) carve((size_t)N_NODES * 128 * 2);   // feat bf16
    bf16*  XB  = (bf16*) carve((size_t)N_NODES * 128 * 2);   // bf16 scratch
    bf16*  MB  = (bf16*) carve((size_t)N_NODES * 160 * 2);   // meta input (padded 160)
    bf16*  in2b  = (bf16*)carve(128 * 128 * 2);
    bf16*  seg2b = (bf16*)carve(128 * 128 * 2);
    bf16*  metab = (bf16*)carve(128 * 160 * 2);
    bf16*  ctrb  = (bf16*)carve((size_t)4 * 128 * 128 * 2);
    bf16*  edgeb = (bf16*)carve((size_t)4 * 14 * 128 * 128 * 2);
    bf16*  ctr2b = (bf16*)carve((size_t)4 * 128 * 128 * 2);
    if (off > ws_size) return;   // insufficient workspace

    const int NB16 = N_NODES / 16;               // 6250
    const int EB16 = (N_EDGES + 15) / 16;        // 6250

    // ---- weight conversion to bf16 ----
    k_cvt<<<(16384 + 255) / 256, 256, 0, stream>>>(in2_W,  in2b,  16384);
    k_cvt<<<(16384 + 255) / 256, 256, 0, stream>>>(seg2_W, seg2b, 16384);
    k_cvt_meta<<<(128 * 160 + 255) / 256, 256, 0, stream>>>(meta_W, metab);
    k_cvt<<<(65536 + 255) / 256, 256, 0, stream>>>(ctr_W,  ctrb,  65536);
    k_cvt<<<(917504 + 255) / 256, 256, 0, stream>>>(edge_W, edgeb, 917504);
    k_cvt<<<(65536 + 255) / 256, 256, 0, stream>>>(ctr2_W, ctr2b, 65536);

    // ---- encoder ----
    k_lin2relu<<<(N_NODES * 128 + 255) / 256, 256, 0, stream>>>(nodes, in1_W, in1_b, XB, 0);
    k_gemm16<<<NB16, 128, 0, stream>>>(XB, 128, in2b, in_g, in_bg, nullptr,
                                       F, nullptr, 0, nullptr, 1);
    k_lin2relu<<<(N_NODES * 128 + 255) / 256, 256, 0, stream>>>(nodes, seg1_W, seg1_b, XB, 2);
    k_gemm16<<<NB16, 128, 0, stream>>>(XB, 128, seg2b, seg_g, seg_bg, F,
                                       nullptr, MB, 160, nullptr, 2);
    k_meta_extra<<<(N_NODES * 32 + 255) / 256, 256, 0, stream>>>(nodes, MB);
    k_gemm16<<<NB16, 128, 0, stream>>>(MB, 160, metab, meta_g, meta_bg, nullptr,
                                       F, FB, 128, R, 3);

    // ---- 4 message-passing layers ----
    for (int i = 0; i < 4; ++i) {
        k_gemm16<<<NB16, 128, 0, stream>>>(FB, 128, ctrb + (size_t)i * 16384,
                                           nullptr, nullptr, nullptr,
                                           T, nullptr, 0, nullptr, 0);
        dim3 eg(EB16, 14);
        k_edges<<<eg, 128, 0, stream>>>(FB, edgeb + (size_t)i * 14 * 16384,
                                        indexes, mask, T);
        k_rownorm_relu<<<NB16, 128, 0, stream>>>(T, norm_g + i * 128,
                                                 norm_bg + i * 128, XB);
        float* dstF = (i < 3) ? F : out;
        k_gemm16<<<NB16, 128, 0, stream>>>(XB, 128, ctr2b + (size_t)i * 16384,
                                           ctr2_g + i * 128, ctr2_bg + i * 128, R,
                                           dstF, FB, 128, R, 4);
    }

    // ---- second tuple output: nodes[:, :2] ----
    k_copy_xy<<<(N_NODES * 2 + 255) / 256, 256, 0, stream>>>(
        nodes, out + (size_t)N_NODES * 128);
}
